// FeedForwardQuantum_65481071404696
// MI455X (gfx1250) — matrix operand order repaired and checked by reading, not compiled, not run on hardware
//
#include <hip/hip_runtime.h>

typedef __attribute__((ext_vector_type(16))) _Float16 v16h;
typedef __attribute__((ext_vector_type(8)))  _Float16 v8h;
typedef __attribute__((ext_vector_type(4)))  _Float16 v4h;
typedef __attribute__((ext_vector_type(8)))  float    v8f;
typedef __attribute__((ext_vector_type(4)))  unsigned int u32x4;

union Frag  { v16h h; u32x4 u[2]; };

#define EDIM 128
#define FDIM 512
#define MT   128          // rows of M per workgroup (8 waves x 16 rows)
#define QS   136          // q LDS row stride in halves (272B, 16B aligned, bank-skewed)
#define HS   520          // h LDS row stride in halves (1040B, 16B aligned, bank-skewed)
// q (34,816B) and h (133,120B) share one region: q is dead once each wave has
// preloaded its q^T B-fragments into VGPRs (guarded by a barrier).
// Total 133,632B < 160KB -> two workgroups co-resident per 320KB WGP, so one
// workgroup's VMEM/cos stage overlaps the other's WMMA stages.
#define LDS_BYTES (512 + MT*HS*2)   // 512 + 133120 = 133632

// ---------------------------------------------------------------------------
// Prologue: convert W1 [512x128] f32 and W2 [128x512] f32 to f16 in workspace.
// ---------------------------------------------------------------------------
__global__ __launch_bounds__(256)
void ffq_convert_weights(const float* __restrict__ W1, const float* __restrict__ W2,
                         _Float16* __restrict__ W1h, _Float16* __restrict__ W2h)
{
    int i = blockIdx.x * 256 + threadIdx.x;      // 0..32767 float4 units
    int which = i >> 14;                         // 0 -> W1, 1 -> W2
    int j = (i & 16383) << 2;
    const float* src = which ? W2 : W1;
    _Float16*    dst = which ? W2h : W1h;
    float4 v = *(const float4*)(src + j);
    v4h o = { (_Float16)v.x, (_Float16)v.y, (_Float16)v.z, (_Float16)v.w };
    *(v4h*)(dst + j) = o;
}

// ---------------------------------------------------------------------------
// Fused, transposed formulation:
//   h^T = W1 . q^T   (A = W1 rows from global/L2, B = q^T rows from LDS)
//   o^T = W2 . h^T   (A = W2 rows from global/L2, B = h^T rows from LDS)
// C-fragments land with lane = m and 8 consecutive f (or e) per lane, so h
// write-back is one ds_store_b128 and output is two coalesced b128 stores.
// Grid: M/128 workgroups of 256 threads (8 wave32s). Each wave owns 16 rows.
// ---------------------------------------------------------------------------
__global__ __launch_bounds__(256)
void ffq_fused(const float* __restrict__ x, const float* __restrict__ theta,
               const _Float16* __restrict__ W1h, const _Float16* __restrict__ W2h,
               float* __restrict__ out)
{
    extern __shared__ char smem[];
    float*    cth   = (float*)smem;                   // 128 f32
    _Float16* q_lds = (_Float16*)(smem + 512);        // MT x QS   (phase 1)
    _Float16* h_lds = (_Float16*)(smem + 512);        // MT x HS   (phase 2, aliases q)

    const int  tid  = threadIdx.x;
    const long row0 = (long)blockIdx.x * MT;

    if (tid < EDIM) cth[tid] = __cosf(theta[tid]);
    __syncthreads();

    // ---- Stage 1: q tile [128 rows x 128 e] f32 -> f16 LDS ------------------
    #pragma unroll
    for (int i = 0; i < 16; ++i) {
        int c  = tid + 256 * i;
        int r  = c >> 5;
        int c4 = (c & 31) << 2;
        const float4 v = *(const float4*)(x + (row0 + r) * EDIM + c4);
        v4h o = { (_Float16)(__cosf(v.x) * cth[c4 + 0]),
                  (_Float16)(__cosf(v.y) * cth[c4 + 1]),
                  (_Float16)(__cosf(v.z) * cth[c4 + 2]),
                  (_Float16)(__cosf(v.w) * cth[c4 + 3]) };
        *(v4h*)(q_lds + r * QS + c4) = o;
    }
    __syncthreads();

    const int wave = tid >> 5;
    const int lane = tid & 31;
    const int l15  = lane & 15;           // A row within tile / B column (= m)
    const int ahk  = (lane >> 4) << 3;    // A-matrix K sub-offset: 0 / 8
    const int bk   = (lane >> 4) << 3;    // B-matrix K sub-offset: 0 / 8
    const int c0   = (lane >> 4) << 3;    // C layout: M(row) = c0 + vgpr_index

    // ---- Preload all B fragments of GEMM1: q^T, K = 128 -> 4 x v16h --------
    // B 32x16 layout: lane holds column m = l15, K = bk..bk+7 and bk+16..bk+23.
    const _Float16* qrow = q_lds + (wave * 16 + l15) * QS;
    Frag bq[4];
    #pragma unroll
    for (int kb = 0; kb < 4; ++kb) {
        bq[kb].u[0] = *(const u32x4*)(qrow + kb * 32 + bk);
        bq[kb].u[1] = *(const u32x4*)(qrow + kb * 32 + bk + 16);
    }
    // All waves must finish reading q before h stores overwrite the region.
    __syncthreads();

    _Float16* hrow = h_lds + (wave * 16 + l15) * HS;   // this lane's m-row of h

    // ---- Stage 2 (GEMM1): h^T[f,m] = relu(W1 . q^T), f16 into LDS ----------
    #pragma unroll 1
    for (int nf = 0; nf < FDIM / 16; ++nf) {
        v8f acc0 = {}, acc1 = {};
        const _Float16* wrow = W1h + (nf * 16 + l15) * EDIM;   // A row f of W1h
        #pragma unroll
        for (int kb = 0; kb < 4; kb += 2) {
            Frag a0, a1;
            a0.u[0] = *(const u32x4*)(wrow + kb * 32 + ahk);
            a0.u[1] = *(const u32x4*)(wrow + kb * 32 + ahk + 16);
            a1.u[0] = *(const u32x4*)(wrow + (kb + 1) * 32 + ahk);
            a1.u[1] = *(const u32x4*)(wrow + (kb + 1) * 32 + ahk + 16);
            acc0 = __builtin_amdgcn_wmma_f32_16x16x32_f16(
                       false, a0.h, false, bq[kb].h, (short)0, acc0, false, false);
            acc1 = __builtin_amdgcn_wmma_f32_16x16x32_f16(
                       false, a1.h, false, bq[kb + 1].h, (short)0, acc1, false, false);
        }
        // relu + pack 8 consecutive f values -> one 16B LDS store
        v8h hv;
        #pragma unroll
        for (int r = 0; r < 8; ++r)
            hv[r] = (_Float16)__builtin_fmaxf(acc0[r] + acc1[r], 0.0f);
        *(v8h*)(hrow + nf * 16 + c0) = hv;
    }
    // Each wave reads back only its own m-rows of h: wave-local LDS RAW,
    // the compiler inserts s_wait_dscnt; no __syncthreads needed.

    // ---- Stage 3 (GEMM2): o^T[e,m] = W2 . h^T, f32 coalesced to global -----
    #pragma unroll 1
    for (int ne = 0; ne < EDIM / 16; ++ne) {
        v8f acc0 = {}, acc1 = {};
        const _Float16* w2row = W2h + (ne * 16 + l15) * FDIM;  // A row e of W2h
        #pragma unroll
        for (int kb = 0; kb < 16; kb += 2) {
            Frag a0, a1, b0, b1;
            a0.u[0] = *(const u32x4*)(w2row + kb * 32 + ahk);
            a0.u[1] = *(const u32x4*)(w2row + kb * 32 + ahk + 16);
            b0.u[0] = *(const u32x4*)(hrow + kb * 32 + bk);
            b0.u[1] = *(const u32x4*)(hrow + kb * 32 + bk + 16);
            a1.u[0] = *(const u32x4*)(w2row + (kb + 1) * 32 + ahk);
            a1.u[1] = *(const u32x4*)(w2row + (kb + 1) * 32 + ahk + 16);
            b1.u[0] = *(const u32x4*)(hrow + (kb + 1) * 32 + bk);
            b1.u[1] = *(const u32x4*)(hrow + (kb + 1) * 32 + bk + 16);
            acc0 = __builtin_amdgcn_wmma_f32_16x16x32_f16(
                       false, a0.h, false, b0.h, (short)0, acc0, false, false);
            acc1 = __builtin_amdgcn_wmma_f32_16x16x32_f16(
                       false, a1.h, false, b1.h, (short)0, acc1, false, false);
        }
        // lane l15 owns output row m; 8 consecutive e values -> 2x b128 stores
        float* orow = out + (row0 + wave * 16 + l15) * EDIM + ne * 16 + c0;
        float4 lo = { acc0[0] + acc1[0], acc0[1] + acc1[1],
                      acc0[2] + acc1[2], acc0[3] + acc1[3] };
        float4 hi = { acc0[4] + acc1[4], acc0[5] + acc1[5],
                      acc0[6] + acc1[6], acc0[7] + acc1[7] };
        *(float4*)(orow)     = lo;
        *(float4*)(orow + 4) = hi;
    }
}

// ---------------------------------------------------------------------------
extern "C" void kernel_launch(void* const* d_in, const int* in_sizes, int n_in,
                              void* d_out, int out_size, void* d_ws, size_t ws_size,
                              hipStream_t stream)
{
    const float* x     = (const float*)d_in[0];   // [32,4096,128]
    const float* theta = (const float*)d_in[1];   // [128]
    const float* W1    = (const float*)d_in[2];   // [512,128]
    const float* W2    = (const float*)d_in[3];   // [128,512]

    _Float16* W1h = (_Float16*)d_ws;              // 65536 halves
    _Float16* W2h = W1h + FDIM * EDIM;            // 65536 halves

    ffq_convert_weights<<<128, 256, 0, stream>>>(W1, W2, W1h, W2h);

    const int M = in_sizes[0] / EDIM;             // 131072 rows
    ffq_fused<<<dim3(M / MT), 256, LDS_BYTES, stream>>>(
        x, theta, W1h, W2h, (float*)d_out);
}
